// InteractionBlock_50714973831856
// MI455X (gfx1250) — matrix operand order repaired and checked by reading, not compiled, not run on hardware
//
#include <hip/hip_runtime.h>

typedef __attribute__((ext_vector_type(16))) _Float16 v16h;
typedef __attribute__((ext_vector_type(8)))  _Float16 h8;
typedef __attribute__((ext_vector_type(4)))  _Float16 h4;
typedef __attribute__((ext_vector_type(8)))  float    v8f;
typedef __attribute__((ext_vector_type(4)))  float    f4;

#define HID 128
#define NRBF 64
#define LDA 72    // row stride (halves) for K=64 staging, %8==0 keeps 16B alignment
#define LDH 136   // row stride (halves) for K=128 staging

__device__ __forceinline__ v16h cat16(h8 lo, h8 hi) {
  return __builtin_shufflevector(lo, hi, 0,1,2,3,4,5,6,7,8,9,10,11,12,13,14,15);
}

// A fragment (16x32 f16). p = &row[m = lane&15] + k0. Lane<16 holds K {0..7,16..23}, lane>=16 holds {8..15,24..31}.
__device__ __forceinline__ v16h loadA_frag(const _Float16* p, int lane) {
  p += (lane < 16) ? 0 : 8;
  h8 lo = *(const h8*)(p);
  h8 hi = *(const h8*)(p + 16);
  return cat16(lo, hi);
}
// B fragment (32x16 f16) from transposed [N][K] storage. p = &col[n = lane&15] + k0.
// Lane<16 holds K {0..7,16..23}, lane>=16 holds {8..15,24..31} (same order as A).
__device__ __forceinline__ v16h loadB_frag(const _Float16* p, int lane) {
  p += (lane < 16) ? 0 : 8;
  h8 lo = *(const h8*)(p);
  h8 hi = *(const h8*)(p + 16);
  return cat16(lo, hi);
}

__device__ __forceinline__ float silu_f(float v) { return v / (1.0f + __expf(-v)); }

__device__ __forceinline__ v8f wmma_f16(v16h a, v16h b, v8f c) {
  return __builtin_amdgcn_wmma_f32_16x16x32_f16(false, a, false, b, (short)0, c, false, false);
}

// ---------------- zero workspace ----------------
__global__ void k_zero(float* __restrict__ p, long long n) {
  long long i = (long long)blockIdx.x * blockDim.x + threadIdx.x;
  if (i < n) p[i] = 0.0f;
}

// ---------------- weight convert: f32 [K][N] -> f16 transposed [N][K] ----------------
__global__ void k_cvt_wT(const float* __restrict__ w, _Float16* __restrict__ wT, int K, int N) {
  int i = blockIdx.x * blockDim.x + threadIdx.x;
  if (i < K * N) {
    int n = i / K;
    int k = i - n * K;
    wT[i] = (_Float16)w[k * N + n];
  }
}

// ---------------- xf = x @ lin1_w (no bias) ----------------
__global__ __launch_bounds__(256) void k_lin1(const float* __restrict__ x,
                                              const _Float16* __restrict__ wT,
                                              float* __restrict__ out, int N) {
  __shared__ _Float16 sIn[HID * LDH];
  int t = threadIdx.x;
  int base = blockIdx.x * 128;
  for (int i = t; i < 128 * 32; i += 256) {      // float4 chunks
    int r = i >> 5, kc = (i & 31) * 4;
    int g = base + r;
    f4 v = {};
    if (g < N) v = *(const f4*)(x + (long long)g * HID + kc);
    *(h4*)(sIn + r * LDH + kc) = __builtin_convertvector(v, h4);
  }
  __syncthreads();
  int wave = t >> 5, lane = t & 31;
  int nlo = lane & 15, rhalf = (lane < 16) ? 0 : 8;
  int rbase = base + wave * 16 + rhalf;
  bool full = (base + 128 <= N);
  const _Float16* aRow = sIn + (wave * 16 + nlo) * LDH;
  #pragma unroll
  for (int n0 = 0; n0 < 8; ++n0) {
    v8f acc = {};
    #pragma unroll
    for (int ks = 0; ks < 4; ++ks)
      acc = wmma_f16(loadA_frag(aRow + ks * 32, lane),
                     loadB_frag(wT + (n0 * 16 + nlo) * HID + ks * 32, lane), acc);
    int n = n0 * 16 + nlo;
    float* po = out + (long long)rbase * HID + n;
    if (full) {
      #pragma unroll
      for (int j = 0; j < 8; ++j) po[(long long)j * HID] = acc[j];
    } else {
      #pragma unroll
      for (int j = 0; j < 8; ++j)
        if (rbase + j < N) po[(long long)j * HID] = acc[j];
    }
  }
}

// ---------------- fused edge kernel: MLP(edge_attr)*C * xf[src] -> atomic scatter ----------------
__global__ __launch_bounds__(256) void k_edge(const float* __restrict__ attr,
                                              const float* __restrict__ ew,
                                              const long long* __restrict__ eidx,
                                              const _Float16* __restrict__ w1T,
                                              const float* __restrict__ b1,
                                              const _Float16* __restrict__ w2T,
                                              const float* __restrict__ b2,
                                              const float* __restrict__ xf,
                                              float* __restrict__ agg,
                                              float* __restrict__ cnt, int E) {
  __shared__ _Float16 sA[128 * LDA];   // edge_attr tile f16 (K=64)
  __shared__ _Float16 sH[128 * LDH];   // silu(h) tile f16 (K=128)
  __shared__ float sC[128];            // cosine cutoff per edge
  __shared__ int   sS[128], sD[128];   // src / dst per edge
  int t = threadIdx.x;
  int base = blockIdx.x * 128;

  // per-edge metadata: cutoff, indices, degree count, L2 prefetch of gather rows
  if (t < 128) {
    int e = base + t;
    float Cv = 0.0f;
    int s = 0, d = 0;
    if (e < E) {
      float w = ew[e];
      Cv = 0.5f * (__cosf(w * 0.6283185307f) + 1.0f);
      Cv = (w < 5.0f) ? Cv : 0.0f;
      s = (int)eidx[e];
      d = (int)eidx[(long long)E + e];
      __builtin_prefetch(xf + (long long)s * HID, 0, 3);   // global_prefetch_b8
      atomicAdd(&cnt[d], 1.0f);                            // degree count (fused)
    }
    sC[t] = Cv; sS[t] = s; sD[t] = d;
  }
  // edge_attr tile -> f16 LDS (float4 loads, h4 stores)
  for (int i = t; i < 128 * 16; i += 256) {
    int r = i >> 4, kc = (i & 15) * 4;
    int g = base + r;
    f4 v = {};
    if (g < E) v = *(const f4*)(attr + (long long)g * NRBF + kc);
    *(h4*)(sA + r * LDA + kc) = __builtin_convertvector(v, h4);
  }
  __syncthreads();

  int wave = t >> 5, lane = t & 31;
  int nlo = lane & 15, rhalf = (lane < 16) ? 0 : 8;

  // GEMM1: h = silu(attr @ w1 + b1)   [128 x 128]
  {
    const _Float16* aRow = sA + (wave * 16 + nlo) * LDA;
    #pragma unroll
    for (int n0 = 0; n0 < 8; ++n0) {
      v8f acc = {};
      #pragma unroll
      for (int ks = 0; ks < 2; ++ks)
        acc = wmma_f16(loadA_frag(aRow + ks * 32, lane),
                       loadB_frag(w1T + (n0 * 16 + nlo) * NRBF + ks * 32, lane), acc);
      int n = n0 * 16 + nlo;
      float bias = b1[n];
      #pragma unroll
      for (int j = 0; j < 8; ++j)
        sH[(wave * 16 + j + rhalf) * LDH + n] = (_Float16)silu_f(acc[j] + bias);
    }
  }
  __syncthreads();

  // GEMM2: W = (h @ w2 + b2) * C ; msg = xf[src]*W ; scatter-add to agg[dst]
  // Out-of-range edges have C==0 and src/dst==0 -> harmless zero adds, no branches.
  {
    const _Float16* aRow = sH + (wave * 16 + nlo) * LDH;
    #pragma unroll
    for (int n0 = 0; n0 < 8; ++n0) {
      v8f acc = {};
      #pragma unroll
      for (int ks = 0; ks < 4; ++ks)
        acc = wmma_f16(loadA_frag(aRow + ks * 32, lane),
                       loadB_frag(w2T + (n0 * 16 + nlo) * HID + ks * 32, lane), acc);
      int n = n0 * 16 + nlo;
      float bias = b2[n];
      #pragma unroll
      for (int j = 0; j < 8; ++j) {
        int el = wave * 16 + j + rhalf;
        float val = (acc[j] + bias) * sC[el];
        float xv = xf[(long long)sS[el] * HID + n];
        atomicAdd(&agg[(long long)sD[el] * HID + n], val * xv);
      }
    }
  }
}

// ---------------- tail1: t = silu((agg/max(cnt,1)) @ lin2_w + lin2_b) ----------------
__global__ __launch_bounds__(256) void k_tail1(const float* __restrict__ agg,
                                               const float* __restrict__ cnt,
                                               const _Float16* __restrict__ wT,
                                               const float* __restrict__ bias,
                                               float* __restrict__ out, int N) {
  __shared__ _Float16 sIn[HID * LDH];
  int t = threadIdx.x;
  int base = blockIdx.x * 128;
  for (int i = t; i < 128 * 32; i += 256) {
    int r = i >> 5, kc = (i & 31) * 4;
    int g = base + r;
    f4 v = {};
    if (g < N) {
      float rcp = 1.0f / fmaxf(cnt[g], 1.0f);
      v = *(const f4*)(agg + (long long)g * HID + kc);
      v *= rcp;
    }
    *(h4*)(sIn + r * LDH + kc) = __builtin_convertvector(v, h4);
  }
  __syncthreads();
  int wave = t >> 5, lane = t & 31;
  int nlo = lane & 15, rhalf = (lane < 16) ? 0 : 8;
  int rbase = base + wave * 16 + rhalf;
  bool full = (base + 128 <= N);
  const _Float16* aRow = sIn + (wave * 16 + nlo) * LDH;
  #pragma unroll
  for (int n0 = 0; n0 < 8; ++n0) {
    v8f acc = {};
    #pragma unroll
    for (int ks = 0; ks < 4; ++ks)
      acc = wmma_f16(loadA_frag(aRow + ks * 32, lane),
                     loadB_frag(wT + (n0 * 16 + nlo) * HID + ks * 32, lane), acc);
    int n = n0 * 16 + nlo;
    float b = bias[n];
    float* po = out + (long long)rbase * HID + n;
    if (full) {
      #pragma unroll
      for (int j = 0; j < 8; ++j) po[(long long)j * HID] = silu_f(acc[j] + b);
    } else {
      #pragma unroll
      for (int j = 0; j < 8; ++j)
        if (rbase + j < N) po[(long long)j * HID] = silu_f(acc[j] + b);
    }
  }
}

// ---------------- tail2: out = t @ lin_w + lin_b ----------------
__global__ __launch_bounds__(256) void k_tail2(const float* __restrict__ tin,
                                               const _Float16* __restrict__ wT,
                                               const float* __restrict__ bias,
                                               float* __restrict__ out, int N) {
  __shared__ _Float16 sIn[HID * LDH];
  int t = threadIdx.x;
  int base = blockIdx.x * 128;
  for (int i = t; i < 128 * 32; i += 256) {
    int r = i >> 5, kc = (i & 31) * 4;
    int g = base + r;
    f4 v = {};
    if (g < N) v = *(const f4*)(tin + (long long)g * HID + kc);
    *(h4*)(sIn + r * LDH + kc) = __builtin_convertvector(v, h4);
  }
  __syncthreads();
  int wave = t >> 5, lane = t & 31;
  int nlo = lane & 15, rhalf = (lane < 16) ? 0 : 8;
  int rbase = base + wave * 16 + rhalf;
  bool full = (base + 128 <= N);
  const _Float16* aRow = sIn + (wave * 16 + nlo) * LDH;
  #pragma unroll
  for (int n0 = 0; n0 < 8; ++n0) {
    v8f acc = {};
    #pragma unroll
    for (int ks = 0; ks < 4; ++ks)
      acc = wmma_f16(loadA_frag(aRow + ks * 32, lane),
                     loadB_frag(wT + (n0 * 16 + nlo) * HID + ks * 32, lane), acc);
    int n = n0 * 16 + nlo;
    float b = bias[n];
    float* po = out + (long long)rbase * HID + n;
    if (full) {
      #pragma unroll
      for (int j = 0; j < 8; ++j) po[(long long)j * HID] = acc[j] + b;
    } else {
      #pragma unroll
      for (int j = 0; j < 8; ++j)
        if (rbase + j < N) po[(long long)j * HID] = acc[j] + b;
    }
  }
}

extern "C" void kernel_launch(void* const* d_in, const int* in_sizes, int n_in,
                              void* d_out, int out_size, void* d_ws, size_t ws_size,
                              hipStream_t stream) {
  const float*      x    = (const float*)d_in[0];
  const long long*  eidx = (const long long*)d_in[1];
  const float*      ew   = (const float*)d_in[2];
  const float*      attr = (const float*)d_in[3];
  const float*      w1   = (const float*)d_in[4];
  const float*      b1   = (const float*)d_in[5];
  const float*      w2   = (const float*)d_in[6];
  const float*      b2   = (const float*)d_in[7];
  const float*      l1w  = (const float*)d_in[8];
  const float*      l2w  = (const float*)d_in[9];
  const float*      l2b  = (const float*)d_in[10];
  const float*      lw   = (const float*)d_in[11];
  const float*      lb   = (const float*)d_in[12];

  const int N = in_sizes[0] / HID;
  const int E = in_sizes[2];

  // workspace layout
  char* ws = (char*)d_ws;
  float* xf  = (float*)ws;                                   // [N,128], reused as tail intermediate
  float* agg = (float*)(ws + (size_t)N * HID * 4);           // [N,128]
  float* cnt = (float*)(ws + (size_t)N * HID * 8);           // [N]
  size_t woff = (((size_t)N * HID * 8 + (size_t)N * 4) + 255) & ~(size_t)255;
  _Float16* w1T = (_Float16*)(ws + woff);                    // [128][64]
  _Float16* w2T = w1T + 128 * NRBF;                          // [128][128]
  _Float16* l1T = w2T + 128 * HID;
  _Float16* l2T = l1T + 128 * HID;
  _Float16* lT  = l2T + 128 * HID;

  // 1) zero agg + cnt (contiguous)
  long long nz = (long long)N * HID + N;
  k_zero<<<(unsigned)((nz + 255) / 256), 256, 0, stream>>>(agg, nz);

  // 2) convert weights to transposed f16
  k_cvt_wT<<<(NRBF * HID + 255) / 256, 256, 0, stream>>>(w1,  w1T, NRBF, HID);
  k_cvt_wT<<<(HID * HID + 255) / 256, 256, 0, stream>>>(w2,  w2T, HID, HID);
  k_cvt_wT<<<(HID * HID + 255) / 256, 256, 0, stream>>>(l1w, l1T, HID, HID);
  k_cvt_wT<<<(HID * HID + 255) / 256, 256, 0, stream>>>(l2w, l2T, HID, HID);
  k_cvt_wT<<<(HID * HID + 255) / 256, 256, 0, stream>>>(lw,  lT,  HID, HID);

  // 3) xf = x @ lin1_w
  k_lin1<<<(N + 127) / 128, 256, 0, stream>>>(x, l1T, xf, N);

  // 4) fused edge MLP + cutoff + gather-multiply-scatter (+ degree counts)
  k_edge<<<(E + 127) / 128, 256, 0, stream>>>(attr, ew, eidx, w1T, b1, w2T, b2, xf, agg, cnt, E);

  // 5) tail: mean -> lin2 -> silu (xf buffer is dead, reuse as intermediate)
  k_tail1<<<(N + 127) / 128, 256, 0, stream>>>(agg, cnt, l2T, l2b, xf, N);

  // 6) out = t @ lin_w + lin_b
  k_tail2<<<(N + 127) / 128, 256, 0, stream>>>(xf, lT, lb, (float*)d_out, N);
}
